// GraphTransformerReactionsGFN_56599079026752
// MI455X (gfx1250) — matrix operand order repaired and checked by reading, not compiled, not run on hardware
//
#include <hip/hip_runtime.h>
#include <math.h>

// ---------------------------------------------------------------------------
// GraphTransformerReactionsGFN for MI455X (gfx1250).
// Dense GEMMs -> v_wmma_f32_16x16x32_bf16 (fp32 activations, bf16 tiles in LDS,
// vectorized b128 staging with an interior fast path).
// Graph gather/scatter/segment-softmax -> bandwidth-lean atomic kernels.
// ---------------------------------------------------------------------------

typedef __attribute__((ext_vector_type(16))) __bf16 v16bf;
typedef __attribute__((ext_vector_type(8)))  float  v8f;

// ---------------- fused GEMM:  C = act(A @ W + bias) [+= if addto] ----------
// A: M x K (row major, lda), W: K x N (row major, ldw), C: M x N (ldc).
// Block tile 128x64, K-step 32. 256 threads = 8 waves; wave w owns rows
// [16w,16w+16) x all 64 cols -> four 16x16 WMMA accumulators per wave.
#define GEMM_BM 128
#define GEMM_BN 64
#define GEMM_BK 32
#define LDS_STRIDE 40   // padded row stride (bf16 elems): 80B rows, 16B-aligned chunks

__global__ __launch_bounds__(256) void gemm_bf16_kernel(
    const float* __restrict__ A, int lda,
    const float* __restrict__ W, int ldw,
    const float* __restrict__ bias,
    float* __restrict__ C, int ldc,
    int M, int N, int K, int act, int addto)
{
    __shared__ __bf16 As[GEMM_BM][LDS_STRIDE];   // As[m][k]
    __shared__ __bf16 Bs[GEMM_BN][LDS_STRIDE];   // Bs[n][k]  (W transposed)

    const int tid  = threadIdx.x;
    const int wave = tid >> 5;
    const int lane = tid & 31;
    const int bm0  = blockIdx.x * GEMM_BM;
    const int bn0  = blockIdx.y * GEMM_BN;

    v8f acc0 = {}, acc1 = {}, acc2 = {}, acc3 = {};

    const bool fullM = (bm0 + GEMM_BM) <= M;

    // fast-path staging coordinates (interior tiles)
    const int ar  = tid >> 1;             // A row 0..127 (2 threads per row)
    const int ac0 = (tid & 1) * 16;       // A col chunk 0 or 16
    const int wkr = tid >> 3;             // W k-row 0..31 (8 threads per row)
    const int wn0 = (tid & 7) * 8;        // W n chunk of 8

    for (int k0 = 0; k0 < K; k0 += GEMM_BK) {
        const bool fullK = (k0 + GEMM_BK) <= K;
        if (fullM && fullK) {
            // prefetch next K tile (global_prefetch_b8)
            if (k0 + GEMM_BK < K) {
                __builtin_prefetch(A + (size_t)(bm0 + ar) * lda + (k0 + GEMM_BK + ac0), 0, 1);
                __builtin_prefetch(W + (size_t)(k0 + GEMM_BK + wkr) * ldw + (bn0 + wn0), 0, 1);
            }
            // ---- A tile: 128x32 floats, 16 per thread, b128 in / b128 out ----
            {
                const float* ap = A + (size_t)(bm0 + ar) * lda + (k0 + ac0);
                float4 f[4];
                f[0] = ((const float4*)ap)[0];
                f[1] = ((const float4*)ap)[1];
                f[2] = ((const float4*)ap)[2];
                f[3] = ((const float4*)ap)[3];
                union { __bf16 h[16]; uint4 u[2]; } pk;
                #pragma unroll
                for (int i = 0; i < 4; ++i) {
                    pk.h[4 * i + 0] = (__bf16)f[i].x;
                    pk.h[4 * i + 1] = (__bf16)f[i].y;
                    pk.h[4 * i + 2] = (__bf16)f[i].z;
                    pk.h[4 * i + 3] = (__bf16)f[i].w;
                }
                *(uint4*)&As[ar][ac0]     = pk.u[0];
                *(uint4*)&As[ar][ac0 + 8] = pk.u[1];
            }
            // ---- W tile: 32x64 floats, 8 per thread, transposed into Bs ----
            {
                const float* wp = W + (size_t)(k0 + wkr) * ldw + (bn0 + wn0);
                float4 f0 = ((const float4*)wp)[0];
                float4 f1 = ((const float4*)wp)[1];
                __bf16 h[8] = { (__bf16)f0.x, (__bf16)f0.y, (__bf16)f0.z, (__bf16)f0.w,
                                (__bf16)f1.x, (__bf16)f1.y, (__bf16)f1.z, (__bf16)f1.w };
                #pragma unroll
                for (int i = 0; i < 8; ++i) Bs[wn0 + i][wkr] = h[i];
            }
        } else {
            // guarded scalar staging (boundary tiles only)
            for (int i = tid; i < GEMM_BM * GEMM_BK; i += 256) {
                int r = i >> 5, cc = i & 31;
                int gm = bm0 + r, gk = k0 + cc;
                float v = (gm < M && gk < K) ? A[(size_t)gm * lda + gk] : 0.0f;
                As[r][cc] = (__bf16)v;
            }
            for (int i = tid; i < GEMM_BN * GEMM_BK; i += 256) {
                int n = i >> 5, cc = i & 31;
                int gk = k0 + cc;
                float v = (gk < K) ? W[(size_t)gk * ldw + (bn0 + n)] : 0.0f;
                Bs[n][cc] = (__bf16)v;
            }
        }
        __syncthreads();

        // A fragment (16x32): lane = M row (lane&15), kbase=(lane>=16)*16,
        // element e -> K = kbase + e   (matches B fragment K order)
        const int am  = lane & 15;
        const int akb = (lane >> 4) << 4;
        v16bf af;
        const __bf16* ap = &As[wave * 16 + am][0];
        #pragma unroll
        for (int e = 0; e < 8; ++e) {
            af[e]     = ap[akb + e];
            af[e + 8] = ap[akb + 8 + e];
        }
        // B fragments (32x16 each): lane<16 -> N=lane,K=e; lane>=16 -> N=lane-16,K=16+e
        const int bn  = lane & 15;
        const int bkb = (lane >> 4) << 4;
        v16bf bf0, bf1, bf2, bf3;
        const __bf16* bp0 = &Bs[ 0 + bn][0];
        const __bf16* bp1 = &Bs[16 + bn][0];
        const __bf16* bp2 = &Bs[32 + bn][0];
        const __bf16* bp3 = &Bs[48 + bn][0];
        #pragma unroll
        for (int e = 0; e < 16; ++e) {
            bf0[e] = bp0[bkb + e];
            bf1[e] = bp1[bkb + e];
            bf2[e] = bp2[bkb + e];
            bf3[e] = bp3[bkb + e];
        }

        acc0 = __builtin_amdgcn_wmma_f32_16x16x32_bf16(false, af, false, bf0, (short)0, acc0, false, false);
        acc1 = __builtin_amdgcn_wmma_f32_16x16x32_bf16(false, af, false, bf1, (short)0, acc1, false, false);
        acc2 = __builtin_amdgcn_wmma_f32_16x16x32_bf16(false, af, false, bf2, (short)0, acc2, false, false);
        acc3 = __builtin_amdgcn_wmma_f32_16x16x32_bf16(false, af, false, bf3, (short)0, acc3, false, false);
        __syncthreads();
    }

    // C/D layout: VGPR j, lane<16 -> M=j, N=lane; lane>=16 -> M=8+j, N=lane-16
    const int cn  = lane & 15;
    const int cm0 = (lane >> 4) << 3;
    #pragma unroll
    for (int t = 0; t < 4; ++t) {
        v8f acc = (t == 0) ? acc0 : (t == 1) ? acc1 : (t == 2) ? acc2 : acc3;
        int gn = bn0 + t * 16 + cn;
        float bv = bias ? bias[gn] : 0.0f;
        #pragma unroll
        for (int j = 0; j < 8; ++j) {
            int gm = bm0 + wave * 16 + cm0 + j;
            if (gm >= M) continue;
            float v = acc[j] + bv;
            if (act == 1)      v = (v > 0.0f) ? v : 0.01f * v;  // leaky_relu
            else if (act == 2) v = fmaxf(v, 0.0f);              // relu
            float* cp = C + (size_t)gm * ldc + gn;
            if (addto) v += *cp;
            *cp = v;
        }
    }
}

// ------------------------- small utility kernels ---------------------------
__global__ void fill_f32_kernel(float* p, float v, long long n) {
    long long i = (long long)blockIdx.x * 256 + threadIdx.x;
    if (i < n) p[i] = v;
}
__global__ void fill_u32_kernel(unsigned* p, unsigned v, long long n) {
    long long i = (long long)blockIdx.x * 256 + threadIdx.x;
    if (i < n) p[i] = v;
}
__global__ void copy_f32_kernel(const float* s, float* d, long long n) {
    long long i = (long long)blockIdx.x * 256 + threadIdx.x;
    if (i < n) d[i] = s[i];
}

// layernorm (no affine), width 128; one wave per row, 4 elems per lane
__global__ __launch_bounds__(256) void ln_kernel(const float* __restrict__ in,
                                                 float* __restrict__ out, int M) {
    int row  = blockIdx.x * 8 + (threadIdx.x >> 5);
    int lane = threadIdx.x & 31;
    if (row >= M) return;
    const float* r = in + (size_t)row * 128;
    float v0 = r[lane], v1 = r[lane + 32], v2 = r[lane + 64], v3 = r[lane + 96];
    float s = v0 + v1 + v2 + v3;
    for (int o = 16; o; o >>= 1) s += __shfl_xor(s, o);
    float mean = s * (1.0f / 128.0f);
    float d0 = v0 - mean, d1 = v1 - mean, d2 = v2 - mean, d3 = v3 - mean;
    float vs = d0 * d0 + d1 * d1 + d2 * d2 + d3 * d3;
    for (int o = 16; o; o >>= 1) vs += __shfl_xor(vs, o);
    float inv = rsqrtf(vs * (1.0f / 128.0f) + 1e-5f);
    float* w = out + (size_t)row * 128;
    w[lane] = d0 * inv; w[lane + 32] = d1 * inv;
    w[lane + 64] = d2 * inv; w[lane + 96] = d3 * inv;
}

// augmented edge list: [edges | node->graph | graph->node | self loops]
__global__ void build_edges_kernel(const int* __restrict__ ei, const int* __restrict__ batch,
                                   int* __restrict__ src, int* __restrict__ dst,
                                   long long Ea, int E, int N) {
    long long i = (long long)blockIdx.x * 256 + threadIdx.x;
    if (i >= Ea) return;
    int s, d;
    if (i < E)              { s = ei[i];            d = ei[E + i]; }
    else if (i < E + N)     { int j = (int)(i - E);         s = j;            d = batch[j] + N; }
    else if (i < E + 2 * N) { int j = (int)(i - E - N);     s = batch[j] + N; d = j; }
    else                    { int j = (int)(i - E - 2 * N); s = j;            d = j; }
    src[i] = s; dst[i] = d;
}

// virtual-edge one-hot features: ae[E .. E+2N) rows = e_p (col0 = 1)
__global__ void ep_kernel(float* ae, int E, int N) {
    long long t = (long long)blockIdx.x * 256 + threadIdx.x;
    long long total = 2LL * N * 128;
    if (t >= total) return;
    long long row = t >> 7; int c = (int)(t & 127);
    ae[((size_t)E + row) * 128 + c] = (c == 0) ? 1.0f : 0.0f;
}

__global__ void cnt_kernel(const int* __restrict__ dst, float* cnt, long long n) {
    long long i = (long long)blockIdx.x * 256 + threadIdx.x;
    if (i < n) atomicAdd(&cnt[dst[i]], 1.0f);
}
__global__ void loop_scatter_kernel(const int* __restrict__ dst, float* ae, long long nE, int loopBase) {
    long long t = (long long)blockIdx.x * 256 + threadIdx.x;
    if (t >= nE * 128) return;
    long long e = t >> 7; int c = (int)(t & 127);
    atomicAdd(ae + ((size_t)loopBase + dst[e]) * 128 + c, ae[t]);
}
__global__ void loop_div_kernel(float* ae, const float* __restrict__ cnt, int loopBase, int Nt) {
    long long t = (long long)blockIdx.x * 256 + threadIdx.x;
    if (t >= (long long)Nt * 128) return;
    long long n = t >> 7; int c = (int)(t & 127);
    ae[((size_t)loopBase + n) * 128 + c] /= fmaxf(cnt[n], 1.0f);
}

// cs[i] = csg[aug_batch(i)]  (linearity of the cs linear layer)
__global__ void gather_cs_kernel(const float* __restrict__ csg, const int* __restrict__ batch,
                                 float* cs, int Nt, int N) {
    long long t = (long long)blockIdx.x * 256 + threadIdx.x;
    if (t >= (long long)Nt * 256) return;
    long long row = t >> 8; int c = (int)(t & 255);
    int g = (row < N) ? batch[row] : (int)(row - N);
    cs[t] = csg[(size_t)g * 256 + c];
}

__global__ void pack_xc_kernel(const float* __restrict__ on, float* xc, int Nt) {
    long long t = (long long)blockIdx.x * 256 + threadIdx.x;
    if (t >= (long long)Nt * 128) return;
    long long row = t >> 7; int c = (int)(t & 127);
    xc[row * 256 + c] = on[t];
}

__global__ void msg_scatter_kernel(const float* __restrict__ on, const float* __restrict__ ae,
                                   const int* __restrict__ src, const int* __restrict__ dst,
                                   float* agg, long long Ea) {
    long long t = (long long)blockIdx.x * 256 + threadIdx.x;
    if (t >= Ea * 128) return;
    long long e = t >> 7; int c = (int)(t & 127);
    int s = src[e], d = dst[e];
    float m = on[(size_t)s * 128 + c] + ae[t];
    m = (m > 0.0f) ? m : 0.0f;
    atomicAdd(agg + (size_t)d * 128 + c, m + 1e-7f);
}

__global__ void add_inplace_kernel(float* a, const float* __restrict__ b, long long n) {
    long long i = (long long)blockIdx.x * 256 + threadIdx.x;
    if (i < n) a[i] += b[i];
}

// ordered-uint encoding for float atomicMax
__device__ __forceinline__ unsigned fenc(float f) {
    unsigned u = __float_as_uint(f);
    return (u & 0x80000000u) ? ~u : (u | 0x80000000u);
}
__device__ __forceinline__ float fdec(unsigned u) {
    return (u & 0x80000000u) ? __uint_as_float(u & 0x7fffffffu) : __uint_as_float(~u);
}

// attention logit per (edge, head): warp dot over 128 dims + segment atomicMax
__global__ __launch_bounds__(256) void attn_logit_kernel(
    const float* __restrict__ q, const float* __restrict__ k, const float* __restrict__ ee,
    const int* __restrict__ src, const int* __restrict__ dst,
    float* logit, unsigned* segmax, long long Ea)
{
    long long w = (long long)blockIdx.x * 8 + (threadIdx.x >> 5);
    int lane = threadIdx.x & 31;
    if (w >= Ea * 2) return;
    long long e = w >> 1; int h = (int)(w & 1);
    int s = src[e], d = dst[e];
    const float* qp = q  + (size_t)d * 256 + h * 128;
    const float* kp = k  + (size_t)s * 256 + h * 128;
    const float* ep = ee + (size_t)e * 256 + h * 128;
    float acc = 0.0f;
    for (int c = lane; c < 128; c += 32) acc += qp[c] * (kp[c] + ep[c]);
    for (int o = 16; o; o >>= 1) acc += __shfl_xor(acc, o);
    if (lane == 0) {
        float l = acc * 0.08838834764831845f;  // 1/sqrt(128)
        logit[e * 2 + h] = l;
        atomicMax(&segmax[(size_t)d * 2 + h], fenc(l));
    }
}

__global__ void attn_expsum_kernel(const float* __restrict__ logit, const int* __restrict__ dst,
                                   const unsigned* __restrict__ segmax,
                                   float* expv, float* segsum, long long Ea) {
    long long t = (long long)blockIdx.x * 256 + threadIdx.x;
    if (t >= Ea * 2) return;
    long long e = t >> 1; int h = (int)(t & 1);
    int d = dst[e];
    float m = fdec(segmax[(size_t)d * 2 + h]);
    float ev = __expf(logit[t] - m);
    expv[t] = ev;
    atomicAdd(&segsum[(size_t)d * 2 + h], ev);
}

__global__ void attn_val_kernel(const float* __restrict__ v, const float* __restrict__ ee,
                                const float* __restrict__ expv, const float* __restrict__ segsum,
                                const int* __restrict__ src, const int* __restrict__ dst,
                                float* tout, long long Ea) {
    long long t = (long long)blockIdx.x * 256 + threadIdx.x;
    if (t >= Ea * 256) return;
    long long e = t >> 8; int c = (int)(t & 255); int h = c >> 7;
    int s = src[e], d = dst[e];
    float alpha = expv[e * 2 + h] / (segsum[(size_t)d * 2 + h] + 1e-16f);
    atomicAdd(tout + (size_t)d * 256 + c, (v[(size_t)s * 256 + c] + ee[t]) * alpha);
}

// o += l_h * cs[:, :128] + cs[:, 128:]
__global__ void oupdate_kernel(float* o, const float* __restrict__ lh,
                               const float* __restrict__ cs, long long n) {
    long long t = (long long)blockIdx.x * 256 + threadIdx.x;
    if (t >= n) return;
    long long row = t >> 7; int c = (int)(t & 127);
    o[t] += lh[t] * cs[row * 256 + c] + cs[row * 256 + 128 + c];
}

__global__ void gpool_kernel(const float* __restrict__ o, const int* __restrict__ batch,
                             float* gsum, float* gcnt, long long n) {
    long long t = (long long)blockIdx.x * 256 + threadIdx.x;
    if (t >= n) return;
    long long row = t >> 7; int c = (int)(t & 127);
    int g = batch[row];
    atomicAdd(gsum + (size_t)g * 128 + c, o[t]);
    if (c == 0) atomicAdd(&gcnt[g], 1.0f);
}

__global__ void glob_kernel(const float* __restrict__ gsum, const float* __restrict__ gcnt,
                            const float* __restrict__ o, float* out2, int G, int N) {
    long long t = (long long)blockIdx.x * 256 + threadIdx.x;
    if (t >= (long long)G * 256) return;
    long long g = t >> 8; int c = (int)(t & 255);
    out2[t] = (c < 128) ? gsum[(size_t)g * 128 + c] / fmaxf(gcnt[g], 1.0f)
                        : o[((size_t)N + g) * 128 + (c - 128)];
}

// ---------------------------------------------------------------------------
extern "C" void kernel_launch(void* const* d_in, const int* in_sizes, int n_in,
                              void* d_out, int out_size, void* d_ws, size_t ws_size,
                              hipStream_t stream)
{
    (void)in_sizes; (void)n_in; (void)out_size; (void)ws_size;
    const int Nn = 100000, Ee = 400000, Gg = 2000;
    const int Nt = Nn + Gg;            // 102000 augmented nodes
    const int E2 = Ee + 2 * Nn;        // edges before self-loops
    const int Ea = E2 + Nt;            // 702000 augmented edges

    const float* x  = (const float*)d_in[0];
    const float* eat= (const float*)d_in[1];
    const int*   ei = (const int*)d_in[2];
    const int*   bt = (const int*)d_in[3];
    const float* cd = (const float*)d_in[4];
    auto F = [&](int i) { return (const float*)d_in[i]; };
    // params flattened in insertion order:
    //  5..10  x2h  (w0,b0,w1,b1,w2,b2)   11..16 e2h   17..22 c2h
    //  per layer l (base 23 + 19*l): gen.w,gen.b, wq.w,wq.b, wk.w,wk.b,
    //  wv.w,wv.b, we, skip.w,skip.b, lin.w,lin.b, ff0.w,ff0.b, ff1.w,ff1.b,
    //  cs.w,cs.b

    float* ws = (float*)d_ws;
    long long off = 0;
    auto alloc = [&](long long n) { float* p = ws + off; off += n; return p; };
    float*    o      = alloc((long long)Nt * 128);
    float*    on     = alloc((long long)Nt * 128);
    float*    agg    = alloc((long long)Nt * 128);   // reused as l_h
    float*    xc     = alloc((long long)Nt * 256);
    float*    q      = alloc((long long)Nt * 256);
    float*    kk     = alloc((long long)Nt * 256);
    float*    vv     = alloc((long long)Nt * 256);
    float*    tout   = alloc((long long)Nt * 256);
    float*    cs     = alloc((long long)Nt * 256);
    float*    cbuf   = alloc((long long)Gg * 128);
    float*    csg    = alloc((long long)Gg * 256);
    float*    aeb    = alloc((long long)Ea * 128);
    float*    eeb    = alloc((long long)Ea * 256);   // also hosts tmpA/tmpB and ffh
    float*    logit  = alloc((long long)Ea * 2);
    float*    expv   = alloc((long long)Ea * 2);
    unsigned* segmax = (unsigned*)alloc((long long)Nt * 2);
    float*    segsum = alloc((long long)Nt * 2);
    float*    cnt    = alloc(Nt);
    float*    gsum   = alloc((long long)Gg * 128);
    float*    gcnt   = alloc(Gg);
    int*      srcv   = (int*)alloc(Ea);
    int*      dstv   = (int*)alloc(Ea);

    float* tmpA = eeb;                         // Ee x 128 ping
    float* tmpB = eeb + (long long)Ee * 128;   // Ee x 128 pong
    float* ffh  = eeb;                         // Nt x 512 (post-attention reuse)

    auto gemm = [&](const float* A, int lda, const float* W, int ldw, const float* bias,
                    float* C, int ldc, int M, int N, int K, int act, int addto) {
        dim3 g((unsigned)((M + GEMM_BM - 1) / GEMM_BM), (unsigned)(N / GEMM_BN));
        gemm_bf16_kernel<<<g, 256, 0, stream>>>(A, lda, W, ldw, bias, C, ldc, M, N, K, act, addto);
    };
    auto nblk = [](long long n) { return dim3((unsigned)((n + 255) / 256)); };
    auto fillf = [&](float* p, float v, long long n) {
        fill_f32_kernel<<<nblk(n), 256, 0, stream>>>(p, v, n);
    };

    // ---------------- input encoders (MLP3, leaky->leaky->linear) ----------
    gemm(x,    72,  F(5),  128, F(6),  tmpA, 128, Nn, 128, 72,  1, 0);
    gemm(tmpA, 128, F(7),  128, F(8),  tmpB, 128, Nn, 128, 128, 1, 0);
    gemm(tmpB, 128, F(9),  128, F(10), o,    128, Nn, 128, 128, 0, 0);

    gemm(cd,   32,  F(17), 128, F(18), tmpA, 128, Gg, 128, 32,  1, 0);
    gemm(tmpA, 128, F(19), 128, F(20), tmpB, 128, Gg, 128, 128, 1, 0);
    gemm(tmpB, 128, F(21), 128, F(22), cbuf, 128, Gg, 128, 128, 0, 0);
    copy_f32_kernel<<<nblk((long long)Gg * 128), 256, 0, stream>>>(
        cbuf, o + (long long)Nn * 128, (long long)Gg * 128);

    gemm(eat,  16,  F(11), 128, F(12), tmpA, 128, Ee, 128, 16,  1, 0);
    gemm(tmpA, 128, F(13), 128, F(14), tmpB, 128, Ee, 128, 128, 1, 0);
    gemm(tmpB, 128, F(15), 128, F(16), aeb,  128, Ee, 128, 128, 0, 0);

    // ---------------- augmented graph + self-loop features ------------------
    build_edges_kernel<<<nblk(Ea), 256, 0, stream>>>(ei, bt, srcv, dstv, Ea, Ee, Nn);
    ep_kernel<<<nblk(2LL * Nn * 128), 256, 0, stream>>>(aeb, Ee, Nn);
    fillf(cnt, 0.0f, Nt);
    fillf(aeb + (long long)E2 * 128, 0.0f, (long long)Nt * 128);
    cnt_kernel<<<nblk(E2), 256, 0, stream>>>(dstv, cnt, E2);
    loop_scatter_kernel<<<nblk((long long)E2 * 128), 256, 0, stream>>>(dstv, aeb, E2, E2);
    loop_div_kernel<<<nblk((long long)Nt * 128), 256, 0, stream>>>(aeb, cnt, E2, Nt);

    // ---------------- transformer layers ------------------------------------
    for (int l = 0; l < 3; ++l) {
        int pb = 23 + l * 19;
        const float *genw = F(pb + 0),  *genb = F(pb + 1);
        const float *wqw  = F(pb + 2),  *wqb  = F(pb + 3);
        const float *wkw  = F(pb + 4),  *wkb  = F(pb + 5);
        const float *wvw  = F(pb + 6),  *wvb  = F(pb + 7);
        const float *wev  = F(pb + 8);
        const float *skw  = F(pb + 9),  *skb  = F(pb + 10);
        const float *linw = F(pb + 11), *linb = F(pb + 12);
        const float *f0w  = F(pb + 13), *f0b  = F(pb + 14);
        const float *f1w  = F(pb + 15), *f1b  = F(pb + 16);
        const float *csw  = F(pb + 17), *csb  = F(pb + 18);

        // cs = lin(c)[aug_batch]  (tiny GEMM + gather)
        gemm(cbuf, 128, csw, 256, csb, csg, 256, Gg, 256, 128, 0, 0);
        gather_cs_kernel<<<nblk((long long)Nt * 256), 256, 0, stream>>>(csg, bt, cs, Nt, Nn);

        ln_kernel<<<(Nt + 7) / 8, 256, 0, stream>>>(o, on, Nt);
        pack_xc_kernel<<<nblk((long long)Nt * 128), 256, 0, stream>>>(on, xc, Nt);

        // message passing:  agg = gen(segsum(relu(on[src]+ae)+eps) + on)
        fillf(agg, 0.0f, (long long)Nt * 128);
        msg_scatter_kernel<<<nblk((long long)Ea * 128), 256, 0, stream>>>(on, aeb, srcv, dstv, agg, Ea);
        add_inplace_kernel<<<nblk((long long)Nt * 128), 256, 0, stream>>>(agg, on, (long long)Nt * 128);
        gemm(agg, 128, genw, 128, genb, xc + 128, 256, Nt, 128, 128, 0, 0);

        // attention projections
        gemm(xc,  256, wqw, 256, wqb,     q,   256, Nt, 256, 256, 0, 0);
        gemm(xc,  256, wkw, 256, wkb,     kk,  256, Nt, 256, 256, 0, 0);
        gemm(xc,  256, wvw, 256, wvb,     vv,  256, Nt, 256, 256, 0, 0);
        gemm(aeb, 128, wev, 256, nullptr, eeb, 256, Ea, 256, 128, 0, 0);

        // segment softmax attention
        fill_u32_kernel<<<nblk((long long)Nt * 2), 256, 0, stream>>>(segmax, 0u, (long long)Nt * 2);
        fillf(segsum, 0.0f, (long long)Nt * 2);
        attn_logit_kernel<<<(unsigned)((Ea * 2LL + 7) / 8), 256, 0, stream>>>(
            q, kk, eeb, srcv, dstv, logit, segmax, Ea);
        attn_expsum_kernel<<<nblk((long long)Ea * 2), 256, 0, stream>>>(
            logit, dstv, segmax, expv, segsum, Ea);
        fillf(tout, 0.0f, (long long)Nt * 256);
        attn_val_kernel<<<nblk((long long)Ea * 256), 256, 0, stream>>>(
            vv, eeb, expv, segsum, srcv, dstv, tout, Ea);

        // tout += skip(xc);  l_h = lin(tout);  o += l_h*cs[:,:128] + cs[:,128:]
        gemm(xc,   256, skw,  256, skb,  tout, 256, Nt, 256, 256, 0, 1);
        gemm(tout, 256, linw, 128, linb, agg,  128, Nt, 128, 256, 0, 0);
        oupdate_kernel<<<nblk((long long)Nt * 128), 256, 0, stream>>>(o, agg, cs, (long long)Nt * 128);

        // feedforward with residual fused into second GEMM
        ln_kernel<<<(Nt + 7) / 8, 256, 0, stream>>>(o, on, Nt);
        gemm(on,  128, f0w, 512, f0b, ffh, 512, Nt, 512, 128, 1, 0);
        gemm(ffh, 512, f1w, 128, f1b, o,   128, Nt, 128, 512, 0, 1);
    }

    // ---------------- outputs ------------------------------------------------
    float* out = (float*)d_out;
    copy_f32_kernel<<<nblk((long long)Nn * 128), 256, 0, stream>>>(o, out, (long long)Nn * 128);
    fillf(gsum, 0.0f, (long long)Gg * 128);
    fillf(gcnt, 0.0f, Gg);
    gpool_kernel<<<nblk((long long)Nn * 128), 256, 0, stream>>>(o, bt, gsum, gcnt, (long long)Nn * 128);
    glob_kernel<<<nblk((long long)Gg * 256), 256, 0, stream>>>(gsum, gcnt, o, out + (long long)Nn * 128, Gg, Nn);
}
